// GraphResBlock2_15487652069469
// MI455X (gfx1250) — matrix operand order repaired and checked by reading, not compiled, not run on hardware
//
#include <hip/hip_runtime.h>

// ---------------------------------------------------------------------------
// GraphResBlock for MI455X (gfx1250): scatter-mean (f32 atomics) + f16 WMMA
// GEMMs (v_wmma_f32_16x16x32_f16) + fused BN/ReLU epilogues.
// Memory-bound workload (~3.5-4 GB moved vs 144 GFLOP); f16 WMMA keeps the
// matrix math far below the HBM time while f32 segment sums keep the means
// exact. A-operand buffers are padded to 32-row multiples so the WMMA K-loop
// has no per-iteration EXEC masking (stores remain guarded).
// ---------------------------------------------------------------------------

typedef __attribute__((ext_vector_type(16))) _Float16 v16h;
typedef __attribute__((ext_vector_type(8)))  _Float16 h8;
typedef __attribute__((ext_vector_type(4)))  _Float16 h4;
typedef __attribute__((ext_vector_type(8)))  float    v8f;

#define NET 7
#define BN_EPS 1e-5f

// ---------------- utility kernels ----------------

__global__ void fill_zero4_k(float* __restrict__ p, long long n4) {
    long long i = (long long)blockIdx.x * blockDim.x + threadIdx.x;
    if (i < n4) ((float4*)p)[i] = make_float4(0.f, 0.f, 0.f, 0.f);
}

__global__ void count_edges_k(const int* __restrict__ row, const int* __restrict__ et,
                              long long E, float* __restrict__ cnt) {
    long long i = (long long)blockIdx.x * blockDim.x + threadIdx.x;
    if (i >= E) return;
    atomicAdd(&cnt[(size_t)row[i] * NET + et[i]], 1.0f);
}

__global__ void cvt_f2h_k(const float* __restrict__ src, _Float16* __restrict__ dst,
                          long long n) {
    long long i = (long long)blockIdx.x * blockDim.x + threadIdx.x;
    if (i < n) dst[i] = (_Float16)src[i];
}

// Wt[n*K + k] = W[k*Cout + n]  (row-major [K][Cout] -> [Cout][K], f32 -> f16)
__global__ void transpose_h_k(const float* __restrict__ W, _Float16* __restrict__ Wt,
                              int K, int Cout) {
    long long i = (long long)blockIdx.x * blockDim.x + threadIdx.x;
    long long total = (long long)K * Cout;
    if (i >= total) return;
    int n = (int)(i / K);
    int k = (int)(i - (long long)n * K);
    Wt[i] = (_Float16)W[(size_t)k * Cout + n];
}

// ---------------- scatter-mean ----------------

// gather f16 neighbor features, f32 atomic accumulate keyed on (dst, edge_type)
__global__ void scatter_add_h_k(const _Float16* __restrict__ src,
                                const int* __restrict__ row, const int* __restrict__ col,
                                const int* __restrict__ et,
                                long long E, int C, float* __restrict__ sums) {
    long long tid = (long long)blockIdx.x * blockDim.x + threadIdx.x;
    const int cpe = C >> 3;                    // 8 halfs (16B) per thread
    long long total = E * cpe;
    if (tid >= total) return;
    long long e = tid / cpe;
    int ch = (int)(tid - e * cpe);
    int rn = row[e], cn = col[e], t = et[e];
    h8 v = ((const h8*)(src + (size_t)cn * C))[ch];
    float* dst = sums + ((size_t)rn * NET + t) * C + (size_t)ch * 8;
#pragma unroll
    for (int i = 0; i < 8; ++i) atomicAdd(dst + i, (float)v[i]);
}

// mean = sums / max(cnt,1), emitted as f16 packed [N][7*C]
__global__ void normalize_h_k(const float* __restrict__ sums, const float* __restrict__ cnt,
                              _Float16* __restrict__ mh, long long nSeg, int C) {
    long long i = (long long)blockIdx.x * blockDim.x + threadIdx.x;
    const int cpe = C >> 2;
    long long total = nSeg * cpe;
    if (i >= total) return;
    long long seg = i / cpe;
    float inv = 1.0f / fmaxf(cnt[seg], 1.0f);
    float4 v = ((const float4*)sums)[i];
    h4 o;
    o[0] = (_Float16)(v.x * inv);
    o[1] = (_Float16)(v.y * inv);
    o[2] = (_Float16)(v.z * inv);
    o[3] = (_Float16)(v.w * inv);
    ((h4*)mh)[i] = o;
}

// ---------------- WMMA GEMM ----------------
// C[rowsN][Cout] = A[rowsPad][K] (f16) x Bt[Cout][K]^T (f16), f32 accumulate.
// Block: 256 threads = 8 waves as 2(M) x 4(N); block tile 32 x Cout(=256).
// Each wave: 16 rows x 64 cols = 4 accumulators, A-fragment reused 4x.
// A is allocated with rows padded to a multiple of 32, so every load is
// in-bounds and the K-loop carries no EXEC masking; only stores are guarded.
__global__ __launch_bounds__(256) void gemm_wmma_f16_k(
    const _Float16* __restrict__ A, const _Float16* __restrict__ Bt,
    float* __restrict__ C, int rowsN, int K, int Cout) {
    const int lane = threadIdx.x & 31;
    const int wave = threadIdx.x >> 5;
    const int mw = wave >> 2;            // 0..1
    const int nw = wave & 3;             // 0..3
    const int l16 = lane & 15;
    const int h   = lane >> 4;           // lane half: 0 or 1

    const int rowA = blockIdx.x * 32 + mw * 16 + l16;   // always < rowsPad
    const _Float16* Arow = A + (size_t)rowA * K;

    v8f acc[4];
#pragma unroll
    for (int s = 0; s < 4; ++s)
#pragma unroll
        for (int i = 0; i < 8; ++i) acc[s][i] = 0.0f;

    const _Float16* Bcol[4];
#pragma unroll
    for (int s = 0; s < 4; ++s)
        Bcol[s] = Bt + (size_t)(nw * 64 + s * 16 + l16) * K;

    for (int k0 = 0; k0 < K; k0 += 32) {
        // warm next A chunk in the WGP cache (global_prefetch_b8)
        __builtin_prefetch(Arow + k0 + 32, 0, 1);

        // A fragment, 16-bit 16x32 layout:
        //   j=0..7  -> K = h*8 + j ;  j=8..15 -> K = 16 + h*8 + (j-8)
        h8 a0 = *(const h8*)(Arow + k0 + h * 8);
        h8 a1 = *(const h8*)(Arow + k0 + 16 + h * 8);
        v16h a;
#pragma unroll
        for (int i = 0; i < 8; ++i) { a[i] = a0[i]; a[8 + i] = a1[i]; }

#pragma unroll
        for (int s = 0; s < 4; ++s) {
            // B fragment, 16-bit 32x16 layout: j=0..7 -> K = h*8 + j ; j=8..15 -> K = 16 + h*8 + (j-8)
            h8 b0 = *(const h8*)(Bcol[s] + k0 + h * 8);
            h8 b1 = *(const h8*)(Bcol[s] + k0 + 16 + h * 8);
            v16h b;
#pragma unroll
            for (int i = 0; i < 8; ++i) { b[i] = b0[i]; b[8 + i] = b1[i]; }
            acc[s] = __builtin_amdgcn_wmma_f32_16x16x32_f16(
                false, a, false, b, (short)0, acc[s], false, false);
        }
    }

    // C/D 32-bit layout: lane<16 -> M = r, lane>=16 -> M = r+8 (r = vgpr idx)
    const int mBase = blockIdx.x * 32 + mw * 16 + h * 8;
#pragma unroll
    for (int s = 0; s < 4; ++s) {
        const int ncol = nw * 64 + s * 16 + l16;
#pragma unroll
        for (int r = 0; r < 8; ++r) {
            const int m = mBase + r;
            if (m < rowsN) C[(size_t)m * Cout + ncol] = acc[s][r];
        }
    }
}

// ---------------- BatchNorm (training-mode batch stats) ----------------

// per-column sum / sumsq into stats[0..Cout) and stats[Cout..2*Cout)
__global__ void colstats_k(const float* __restrict__ X, int rowsN, int Cout,
                           float* __restrict__ stats) {
    const int c = threadIdx.x;      // blockDim.x == Cout
    float s = 0.f, q = 0.f;
    for (int r = blockIdx.x; r < rowsN; r += gridDim.x) {
        float v = X[(size_t)r * Cout + c];
        s += v; q += v * v;
    }
    atomicAdd(&stats[c], s);
    atomicAdd(&stats[Cout + c], q);
}

// y = relu(bn(x)) emitted as f16 (feeds the second scatter-mean)
__global__ void bn_relu_h_k(const float* __restrict__ X, _Float16* __restrict__ Y,
                            long long total, int rowsN, int Cout,
                            const float* __restrict__ stats,
                            const float* __restrict__ gamma, const float* __restrict__ beta) {
    long long i = (long long)blockIdx.x * blockDim.x + threadIdx.x;
    if (i >= total) return;
    int c = (int)(i % Cout);
    float invN = 1.0f / (float)rowsN;
    float m = stats[c] * invN;
    float var = stats[Cout + c] * invN - m * m;
    float rs = rsqrtf(var + BN_EPS);
    float v = (X[i] - m) * rs * gamma[c] + beta[c];
    Y[i] = (_Float16)fmaxf(v, 0.0f);
}

// out = relu(bn_b(x2) + bn_1(sc)); sc lives in d_out (read then overwrite)
__global__ void final_k(const float* __restrict__ x2, float* __restrict__ outsc,
                        long long total, int rowsN, int Cout,
                        const float* __restrict__ statsB, const float* __restrict__ statsS,
                        const float* __restrict__ gb, const float* __restrict__ bb,
                        const float* __restrict__ g1, const float* __restrict__ b1) {
    long long i = (long long)blockIdx.x * blockDim.x + threadIdx.x;
    if (i >= total) return;
    int c = (int)(i % Cout);
    float invN = 1.0f / (float)rowsN;
    float mB = statsB[c] * invN;
    float vB = statsB[Cout + c] * invN - mB * mB;
    float mS = statsS[c] * invN;
    float vS = statsS[Cout + c] * invN - mS * mS;
    float a = (x2[i]    - mB) * rsqrtf(vB + BN_EPS) * gb[c] + bb[c];
    float s = (outsc[i] - mS) * rsqrtf(vS + BN_EPS) * g1[c] + b1[c];
    outsc[i] = fmaxf(a + s, 0.0f);
}

// ---------------- driver ----------------

static inline unsigned nblk(long long n, int t) { return (unsigned)((n + t - 1) / t); }

extern "C" void kernel_launch(void* const* d_in, const int* in_sizes, int n_in,
                              void* d_out, int out_size, void* d_ws, size_t ws_size,
                              hipStream_t stream) {
    const float* x    = (const float*)d_in[0];
    const int*   ei   = (const int*)  d_in[1];
    const int*   et   = (const int*)  d_in[2];
    // d_in[3] = node_type (ignored: n_node_type == 0 path)
    const float* Wa   = (const float*)d_in[4];
    const float* ga   = (const float*)d_in[5];
    const float* ba   = (const float*)d_in[6];
    const float* Wb   = (const float*)d_in[7];
    const float* gb   = (const float*)d_in[8];
    const float* bb   = (const float*)d_in[9];
    const float* W1   = (const float*)d_in[10];
    const float* g1   = (const float*)d_in[11];
    const float* b1   = (const float*)d_in[12];

    const long long N    = in_sizes[3];
    const long long E    = in_sizes[2];
    const int       CIN  = (int)((long long)in_sizes[0] / N);   // 128
    const int       COUT = in_sizes[5];                          // 256
    const long long NPAD = (N + 31) & ~31LL;                     // WMMA row padding
    const int*      row  = ei;
    const int*      col  = ei + E;
    float*          out  = (float*)d_out;

    // ---- workspace carve-up (f32 regions first, then f16; all 16B aligned) ----
    float* f = (float*)d_ws;
    float* sums  = f;  f += (size_t)N * NET * COUT;   // reused: phase A uses N*7*CIN
    float* cnt   = f;  f += (size_t)N * NET;
    float* x1pre = f;  f += (size_t)N * COUT;
    float* x2pre = f;  f += (size_t)N * COUT;
    float* stats = f;  f += 8 * (size_t)COUT;         // [A: 2C][B: 2C][S: 2C]
    _Float16* hh   = (_Float16*)f;
    _Float16* meanH = hh;  hh += (size_t)NPAD * NET * COUT;  // padded rows: loads always legal
    _Float16* x1h   = hh;  hh += (size_t)N * COUT;
    _Float16* xh    = hh;  hh += (size_t)NPAD * CIN;         // padded rows for shortcut GEMM
    _Float16* WtA   = hh;  hh += (size_t)NET * CIN * COUT;
    _Float16* WtB   = hh;  hh += (size_t)NET * COUT * COUT;
    _Float16* Wt1   = hh;  hh += (size_t)CIN * COUT;

    float* statsA = stats;
    float* statsB = stats + 2 * COUT;
    float* statsS = stats + 4 * COUT;

    const int T = 256;

    // --- shared prep ---
    fill_zero4_k<<<nblk(N * NET / 4, T), T, 0, stream>>>(cnt, N * NET / 4);
    fill_zero4_k<<<nblk(6LL * COUT / 4, T), T, 0, stream>>>(stats, 6LL * COUT / 4);
    count_edges_k<<<nblk(E, T), T, 0, stream>>>(row, et, E, cnt);
    cvt_f2h_k<<<nblk(N * CIN, T), T, 0, stream>>>(x, xh, N * CIN);
    transpose_h_k<<<nblk((long long)NET * CIN * COUT, T), T, 0, stream>>>(Wa, WtA, NET * CIN, COUT);
    transpose_h_k<<<nblk((long long)NET * COUT * COUT, T), T, 0, stream>>>(Wb, WtB, NET * COUT, COUT);
    transpose_h_k<<<nblk((long long)CIN * COUT, T), T, 0, stream>>>(W1, Wt1, CIN, COUT);

    // --- conv A: scatter-mean(x) -> GEMM(Wa) -> BN -> ReLU (f16 out) ---
    fill_zero4_k<<<nblk((long long)N * NET * CIN / 4, T), T, 0, stream>>>(sums, (long long)N * NET * CIN / 4);
    scatter_add_h_k<<<nblk(E * (CIN / 8), T), T, 0, stream>>>(xh, row, col, et, E, CIN, sums);
    normalize_h_k<<<nblk((long long)N * NET * CIN / 4, T), T, 0, stream>>>(sums, cnt, meanH, N * NET, CIN);
    gemm_wmma_f16_k<<<nblk(N, 32), 256, 0, stream>>>(meanH, WtA, x1pre, (int)N, NET * CIN, COUT);
    colstats_k<<<512, COUT, 0, stream>>>(x1pre, (int)N, COUT, statsA);
    bn_relu_h_k<<<nblk(N * COUT, T), T, 0, stream>>>(x1pre, x1h, N * COUT, (int)N, COUT, statsA, ga, ba);

    // --- conv B: scatter-mean(x1) -> GEMM(Wb) ---
    fill_zero4_k<<<nblk((long long)N * NET * COUT / 4, T), T, 0, stream>>>(sums, (long long)N * NET * COUT / 4);
    scatter_add_h_k<<<nblk(E * (COUT / 8), T), T, 0, stream>>>(x1h, row, col, et, E, COUT, sums);
    normalize_h_k<<<nblk((long long)N * NET * COUT / 4, T), T, 0, stream>>>(sums, cnt, meanH, N * NET, COUT);
    gemm_wmma_f16_k<<<nblk(N, 32), 256, 0, stream>>>(meanH, WtB, x2pre, (int)N, NET * COUT, COUT);

    // --- shortcut: x @ W1 (pre-BN result staged in d_out) ---
    gemm_wmma_f16_k<<<nblk(N, 32), 256, 0, stream>>>(xh, Wt1, out, (int)N, CIN, COUT);

    // --- stats + fused BN/BN/residual/ReLU epilogue ---
    colstats_k<<<512, COUT, 0, stream>>>(x2pre, (int)N, COUT, statsB);
    colstats_k<<<512, COUT, 0, stream>>>(out, (int)N, COUT, statsS);
    final_k<<<nblk(N * COUT, T), T, 0, stream>>>(x2pre, out, N * COUT, (int)N, COUT,
                                                 statsB, statsS, gb, bb, g1, b1);
}
